// MEGNet_62251255989045
// MI455X (gfx1250) — matrix operand order repaired and checked by reading, not compiled, not run on hardware
//
#include <hip/hip_runtime.h>
#include <hip/hip_bf16.h>

typedef __attribute__((ext_vector_type(16))) _Float16 v16h;
typedef __attribute__((ext_vector_type(8)))  float    v8f;

#define N_NODES_C 100000
#define N_EDGES_C 1200000
#define NG 64
#define CUTOFF_F 6.0f

// ---------------- WMMA helpers (CDNA5 wave32 fragment layouts) ----------------

__device__ __forceinline__ v8f wmma32(v16h a, v16h b, v8f c) {
  // D = A(16x32 f16) * B(32x16 f16) + C(16x16 f32)
  return __builtin_amdgcn_wmma_f32_16x16x32_f16(false, a, false, b, (short)0, c, false, false);
}

// A fragment from row-major LDS [16][ld] f16. lane l holds row l&15,
// K = e + (e<8?0:8) + ((l>>4)?8:0)  (ISA 05_wmma 16-bit A 16x32 layout)
__device__ __forceinline__ v16h loadA(const _Float16* base, int ld, int lane) {
  int row  = lane & 15;
  int kofs = (lane & 16) ? 8 : 0;
  const _Float16* p = base + row * ld + kofs;
  union { uint4 u[2]; v16h v; } t;
  t.u[0] = *(const uint4*)(p);       // K = kofs .. kofs+7
  t.u[1] = *(const uint4*)(p + 16);  // K = kofs+16 .. kofs+23
  return t.v;
}

// B fragment from row-major weights W[32][ld] f16 (out = x @ W^T):
// lane l holds column n0+(l&15); K = e + (e<8?0:8) + ((l>>4)?8:0)
__device__ __forceinline__ v16h loadB(const _Float16* w, int ld, int n0, int lane) {
  int n    = n0 + (lane & 15);
  int kofs = (lane & 16) ? 8 : 0;
  const _Float16* p = w + n * ld + kofs;
  union { uint4 u[2]; v16h v; } t;
  t.u[0] = *(const uint4*)(p);
  t.u[1] = *(const uint4*)(p + 16);
  return t.v;
}

// Store D (C-layout) back to row-major LDS [16][ld] f16 with bias+ReLU.
__device__ __forceinline__ void storeAct(_Float16* dst, int ld, int n0,
                                         v8f d, const float* __restrict__ bias, int lane) {
  int n  = n0 + (lane & 15);
  int mb = (lane & 16) ? 8 : 0;
  float b = bias[n];
#pragma unroll
  for (int r = 0; r < 8; ++r) {
    float v = d[r] + b;
    v = fmaxf(v, 0.0f);
    dst[(mb + r) * ld + n] = (_Float16)v;
  }
}

__device__ __forceinline__ void atomicMaxF(int* addr, float val) {
  int old = *(volatile int*)addr;
  while (__int_as_float(old) < val) {
    int prev = atomicCAS(addr, old, __float_as_int(val));
    if (prev == old) break;
    old = prev;
  }
}

// ---------------- prep: fp32 -> fp16 weight / feature staging ----------------

__global__ void k_prep(const float* geW0, const float* geW1, const float* geW2,
                       const float* gnW0, const float* gnW1, const float* gnW2,
                       const float* gf,
                       _Float16* hgeW0, _Float16* hgeW1, _Float16* hgeW2,
                       _Float16* hgnW0, _Float16* hgnW1, _Float16* hgnW2,
                       _Float16* uh) {
  int t = threadIdx.x, s = blockDim.x;
  for (int i = t; i < 32 * 128; i += s) hgeW0[i] = (_Float16)geW0[i];
  for (int i = t; i < 32 * 96; i += s)  hgnW0[i] = (_Float16)gnW0[i];
  for (int i = t; i < 32 * 32; i += s) {
    hgeW1[i] = (_Float16)geW1[i]; hgeW2[i] = (_Float16)geW2[i];
    hgnW1[i] = (_Float16)gnW1[i]; hgnW2[i] = (_Float16)gnW2[i];
  }
  for (int i = t; i < NG * 32; i += s) uh[i] = (_Float16)gf[i];
}

__global__ void k_embed(const float* __restrict__ emb, const int* __restrict__ z,
                        const int* __restrict__ batch, _Float16* __restrict__ xh,
                        float* __restrict__ ncnt) {
  long t = blockIdx.x * (long)blockDim.x + threadIdx.x;
  long s = (long)gridDim.x * blockDim.x;
  for (long i = t; i < (long)N_NODES_C * 32; i += s) {
    int node = (int)(i >> 5), f = (int)(i & 31);
    xh[i] = (_Float16)emb[z[node] * 32 + f];
    if (f == 0) atomicAdd(&ncnt[batch[node]], 1.0f);
  }
}

// ---------------- fused edge pipeline: smearing + concat + MLP + scatter ------

__launch_bounds__(256)
__global__ void k_edge_mlp(const float* __restrict__ pos, const int* __restrict__ ei,
                           const int* __restrict__ batch,
                           const _Float16* __restrict__ xh, const _Float16* __restrict__ uh,
                           const _Float16* __restrict__ w0g, const _Float16* __restrict__ w1g,
                           const _Float16* __restrict__ w2g,
                           const float* __restrict__ b0, const float* __restrict__ b1,
                           const float* __restrict__ b2,
                           float* __restrict__ edge_resid,
                           float* __restrict__ aggsum, float* __restrict__ deg,
                           float* __restrict__ esum, float* __restrict__ ecnt,
                           int* __restrict__ eb_buf) {
  __shared__ __align__(16) _Float16 sw0[32 * 128];
  __shared__ __align__(16) _Float16 sw1[32 * 32];
  __shared__ __align__(16) _Float16 sw2[32 * 32];
  __shared__ __align__(16) _Float16 stage[8][16 * 128];
  __shared__ int   sdst[8][16];
  __shared__ int   sgid[8][16];
  __shared__ float s_esum[NG * 32];
  __shared__ float s_ecnt[NG];

  int tid = threadIdx.x;
  for (int i = tid; i < 32 * 128; i += 256) sw0[i] = w0g[i];
  for (int i = tid; i < 32 * 32; i += 256) { sw1[i] = w1g[i]; sw2[i] = w2g[i]; }
  for (int i = tid; i < NG * 32; i += 256) s_esum[i] = 0.f;
  for (int i = tid; i < NG; i += 256) s_ecnt[i] = 0.f;
  __syncthreads();

  const int lane = tid & 31, wv = tid >> 5;
  const int gw = blockIdx.x * 8 + wv, nw = gridDim.x * 8;
  const int ntiles = N_EDGES_C / 16;
  const float delta = CUTOFF_F / 31.0f;
  const float coeff = -0.5f / (delta * delta);

  for (int tile = gw; tile < ntiles; tile += nw) {
    _Float16* st = stage[wv];
    const int m = lane & 15, half = lane >> 4;
    const int ge = tile * 16 + m;
    const int s = ei[ge];
    const int d = ei[N_EDGES_C + ge];
    if (half == 0) {
      const uint4* ps = (const uint4*)(xh + (size_t)s * 32);
      const uint4* pd = (const uint4*)(xh + (size_t)d * 32);
      uint4* o = (uint4*)(st + m * 128);
      o[0] = ps[0]; o[1] = ps[1]; o[2] = ps[2]; o[3] = ps[3];
      o[4] = pd[0]; o[5] = pd[1]; o[6] = pd[2]; o[7] = pd[3];
      sdst[wv][m] = d;
      atomicAdd(&deg[d], 1.0f);
    } else {
      float dx = pos[s * 3 + 0] - pos[d * 3 + 0];
      float dy = pos[s * 3 + 1] - pos[d * 3 + 1];
      float dz = pos[s * 3 + 2] - pos[d * 3 + 2];
      float dist = sqrtf(dx * dx + dy * dy + dz * dz);
#pragma unroll
      for (int j = 0; j < 32; ++j) {
        float td = dist - delta * (float)j;
        st[m * 128 + 64 + j] = (_Float16)__expf(coeff * td * td);
      }
      int g = batch[s];
      sgid[wv][m] = g;
      eb_buf[ge] = g;
      atomicAdd(&s_ecnt[g], 1.0f);
      const uint4* pu = (const uint4*)(uh + g * 32);
      uint4* o = (uint4*)(st + m * 128 + 96);
      o[0] = pu[0]; o[1] = pu[1]; o[2] = pu[2]; o[3] = pu[3];
    }
    __builtin_amdgcn_wave_barrier();

    // layer 0: [16x128] @ [128x32]
    v8f c00 = {}, c01 = {};
#pragma unroll
    for (int kc = 0; kc < 4; ++kc) {
      v16h a = loadA(st + kc * 32, 128, lane);
      c00 = wmma32(a, loadB(sw0 + kc * 32, 128, 0, lane), c00);
      c01 = wmma32(a, loadB(sw0 + kc * 32, 128, 16, lane), c01);
    }
    __builtin_amdgcn_wave_barrier();
    storeAct(st, 128, 0, c00, b0, lane);
    storeAct(st, 128, 16, c01, b0, lane);
    __builtin_amdgcn_wave_barrier();

    // layer 1
    v8f c10 = {}, c11 = {};
    {
      v16h a = loadA(st, 128, lane);
      c10 = wmma32(a, loadB(sw1, 32, 0, lane), c10);
      c11 = wmma32(a, loadB(sw1, 32, 16, lane), c11);
    }
    __builtin_amdgcn_wave_barrier();
    storeAct(st + 32, 128, 0, c10, b1, lane);
    storeAct(st + 32, 128, 16, c11, b1, lane);
    __builtin_amdgcn_wave_barrier();

    // layer 2 (no ReLU)
    v8f c20 = {}, c21 = {};
    {
      v16h a = loadA(st + 32, 128, lane);
      c20 = wmma32(a, loadB(sw2, 32, 0, lane), c20);
      c21 = wmma32(a, loadB(sw2, 32, 16, lane), c21);
    }
    __builtin_amdgcn_wave_barrier();

    const int mb = half ? 8 : 0;
#pragma unroll
    for (int nt = 0; nt < 2; ++nt) {
      int n = nt * 16 + (lane & 15);
      float bb = b2[n];
#pragma unroll
      for (int r = 0; r < 8; ++r) {
        int mm = mb + r;
        int gge = tile * 16 + mm;
        float enew = (nt ? c21[r] : c20[r]) + bb;
        float ea = (float)st[mm * 128 + 64 + n];  // gaussian edge_attr (residual)
        edge_resid[(size_t)gge * 32 + n] = ea + enew;
        atomicAdd(&aggsum[(size_t)sdst[wv][mm] * 32 + n], enew);
        atomicAdd(&s_esum[sgid[wv][mm] * 32 + n], enew);
      }
    }
    __builtin_amdgcn_wave_barrier();
  }
  __syncthreads();
  for (int i = tid; i < NG * 32; i += 256) atomicAdd(&esum[i], s_esum[i]);
  for (int i = tid; i < NG; i += 256) atomicAdd(&ecnt[i], s_ecnt[i]);
}

__global__ void k_div(float* __restrict__ aggsum, const float* __restrict__ deg) {
  long t = blockIdx.x * (long)blockDim.x + threadIdx.x;
  long s = (long)gridDim.x * blockDim.x;
  for (long i = t; i < (long)N_NODES_C * 32; i += s)
    aggsum[i] = aggsum[i] / fmaxf(deg[i >> 5], 1.0f);
}

// ---------------- fused node pipeline -----------------------------------------

__launch_bounds__(256)
__global__ void k_node_mlp(const float* __restrict__ emb, const int* __restrict__ z,
                           const int* __restrict__ batch,
                           const float* __restrict__ agg, const _Float16* __restrict__ uh,
                           const _Float16* __restrict__ w0g, const _Float16* __restrict__ w1g,
                           const _Float16* __restrict__ w2g,
                           const float* __restrict__ b0, const float* __restrict__ b1,
                           const float* __restrict__ b2,
                           float* __restrict__ x_resid, float* __restrict__ xsum) {
  __shared__ __align__(16) _Float16 sw0[32 * 96];
  __shared__ __align__(16) _Float16 sw1[32 * 32];
  __shared__ __align__(16) _Float16 sw2[32 * 32];
  __shared__ __align__(16) _Float16 stage[8][16 * 96];
  __shared__ int szi[8][16];
  __shared__ int sg[8][16];

  int tid = threadIdx.x;
  for (int i = tid; i < 32 * 96; i += 256) sw0[i] = w0g[i];
  for (int i = tid; i < 32 * 32; i += 256) { sw1[i] = w1g[i]; sw2[i] = w2g[i]; }
  __syncthreads();

  const int lane = tid & 31, wv = tid >> 5;
  const int gw = blockIdx.x * 8 + wv, nw = gridDim.x * 8;
  const int ntiles = N_NODES_C / 16;

  for (int tile = gw; tile < ntiles; tile += nw) {
    _Float16* st = stage[wv];
    const int m = lane & 15, half = lane >> 4;
    const int node = tile * 16 + m;
    if (half == 0) {
      int zi = z[node];
      szi[wv][m] = zi;
#pragma unroll
      for (int j = 0; j < 32; ++j) st[m * 96 + j] = (_Float16)emb[zi * 32 + j];
#pragma unroll
      for (int j = 0; j < 32; ++j) st[m * 96 + 32 + j] = (_Float16)agg[(size_t)node * 32 + j];
    } else {
      int g = batch[node];
      sg[wv][m] = g;
      const uint4* pu = (const uint4*)(uh + g * 32);
      uint4* o = (uint4*)(st + m * 96 + 64);
      o[0] = pu[0]; o[1] = pu[1]; o[2] = pu[2]; o[3] = pu[3];
    }
    __builtin_amdgcn_wave_barrier();

    v8f c00 = {}, c01 = {};
#pragma unroll
    for (int kc = 0; kc < 3; ++kc) {
      v16h a = loadA(st + kc * 32, 96, lane);
      c00 = wmma32(a, loadB(sw0 + kc * 32, 96, 0, lane), c00);
      c01 = wmma32(a, loadB(sw0 + kc * 32, 96, 16, lane), c01);
    }
    __builtin_amdgcn_wave_barrier();
    storeAct(st, 96, 0, c00, b0, lane);
    storeAct(st, 96, 16, c01, b0, lane);
    __builtin_amdgcn_wave_barrier();

    v8f c10 = {}, c11 = {};
    {
      v16h a = loadA(st, 96, lane);
      c10 = wmma32(a, loadB(sw1, 32, 0, lane), c10);
      c11 = wmma32(a, loadB(sw1, 32, 16, lane), c11);
    }
    __builtin_amdgcn_wave_barrier();
    storeAct(st + 32, 96, 0, c10, b1, lane);
    storeAct(st + 32, 96, 16, c11, b1, lane);
    __builtin_amdgcn_wave_barrier();

    v8f c20 = {}, c21 = {};
    {
      v16h a = loadA(st + 32, 96, lane);
      c20 = wmma32(a, loadB(sw2, 32, 0, lane), c20);
      c21 = wmma32(a, loadB(sw2, 32, 16, lane), c21);
    }
    __builtin_amdgcn_wave_barrier();

    const int mb = half ? 8 : 0;
#pragma unroll
    for (int nt = 0; nt < 2; ++nt) {
      int n = nt * 16 + (lane & 15);
      float bb = b2[n];
#pragma unroll
      for (int r = 0; r < 8; ++r) {
        int mm = mb + r;
        int nd = tile * 16 + mm;
        float xnew = (nt ? c21[r] : c20[r]) + bb;
        float xv = emb[szi[wv][mm] * 32 + n];
        x_resid[(size_t)nd * 32 + n] = xv + xnew;
        atomicAdd(&xsum[sg[wv][mm] * 32 + n], xnew);
      }
    }
    __builtin_amdgcn_wave_barrier();
  }
}

// ---------------- global MLP (tiny, f32) ---------------------------------------

__global__ void k_global_mlp(const float* __restrict__ gf,
                             const float* __restrict__ xsum, const float* __restrict__ ncnt,
                             const float* __restrict__ esum, const float* __restrict__ ecnt,
                             const float* W0, const float* B0, const float* W1, const float* B1,
                             const float* W2, const float* B2, float* __restrict__ u_resid) {
  int g = threadIdx.x;
  if (g >= NG) return;
  float uin[96];
  float nc = fmaxf(ncnt[g], 1.f), ec = fmaxf(ecnt[g], 1.f);
  for (int j = 0; j < 32; ++j) {
    uin[j]      = xsum[g * 32 + j] / nc;
    uin[32 + j] = esum[g * 32 + j] / ec;
    uin[64 + j] = gf[g * 32 + j];
  }
  float h1[32], h2[32];
  for (int o = 0; o < 32; ++o) {
    float a = B0[o];
    for (int j = 0; j < 96; ++j) a += W0[o * 96 + j] * uin[j];
    h1[o] = fmaxf(a, 0.f);
  }
  for (int o = 0; o < 32; ++o) {
    float a = B1[o];
    for (int j = 0; j < 32; ++j) a += W1[o * 32 + j] * h1[j];
    h2[o] = fmaxf(a, 0.f);
  }
  for (int o = 0; o < 32; ++o) {
    float a = B2[o];
    for (int j = 0; j < 32; ++j) a += W2[o * 32 + j] * h2[j];
    u_resid[g * 32 + o] = gf[g * 32 + o] + a;
  }
}

// ---------------- Set2Set ------------------------------------------------------

__global__ void k_s2s_lstm(const float* __restrict__ Wih, const float* __restrict__ Whh,
                           const float* __restrict__ bih, const float* __restrict__ bhh,
                           float* __restrict__ h, float* __restrict__ c,
                           float* __restrict__ rsum, float* __restrict__ denom,
                           float* __restrict__ emax, int step) {
  __shared__ float sq[NG * 64];
  __shared__ float sz[NG * 128];
  int t = threadIdx.x;
  for (int i = t; i < NG * 64; i += 256) {
    int row = i >> 6, col = i & 63;
    float v;
    if (step == 0) v = 0.f;
    else if (col < 32) v = h[row * 32 + col];
    else v = rsum[row * 32 + (col - 32)] / fmaxf(denom[row], 1e-16f);
    sq[i] = v;
  }
  if (step == 0)
    for (int i = t; i < NG * 32; i += 256) { h[i] = 0.f; c[i] = 0.f; }
  __syncthreads();
  for (int i = t; i < NG * 128; i += 256) {
    int row = i >> 7, col = i & 127;
    float a = bih[col] + bhh[col];
    const float* wi = Wih + col * 64;
    const float* qq = sq + row * 64;
    for (int j = 0; j < 64; ++j) a += qq[j] * wi[j];
    const float* wh = Whh + col * 32;
    const float* hh = h + row * 32;
    for (int j = 0; j < 32; ++j) a += hh[j] * wh[j];
    sz[i] = a;
  }
  __syncthreads();
  for (int i = t; i < NG * 32; i += 256) {
    int row = i >> 5, j = i & 31;
    float zi = sz[row * 128 + j];
    float zf = sz[row * 128 + 32 + j];
    float zg = sz[row * 128 + 64 + j];
    float zo = sz[row * 128 + 96 + j];
    float ig = 1.f / (1.f + __expf(-zi));
    float fg = 1.f / (1.f + __expf(-zf));
    float og = 1.f / (1.f + __expf(-zo));
    float cn = fg * c[i] + ig * tanhf(zg);
    c[i] = cn;
    h[i] = og * tanhf(cn);
  }
  // reset per-step accumulators for the upcoming attention passes
  for (int i = t; i < NG * 32; i += 256) rsum[i] = 0.f;
  for (int i = t; i < NG; i += 256) { denom[i] = 0.f; emax[i] = -__builtin_inff(); }
}

__global__ void k_s2s_pass1(const float* __restrict__ x, const int* __restrict__ seg, int n,
                            const float* __restrict__ h, float* __restrict__ e_buf,
                            float* __restrict__ emax) {
  __shared__ int smax[NG];
  int t = threadIdx.x;
  for (int i = t; i < NG; i += blockDim.x) smax[i] = __float_as_int(-__builtin_inff());
  __syncthreads();
  int stride = gridDim.x * blockDim.x;
  for (int i = blockIdx.x * blockDim.x + t; i < n; i += stride) {
    int g = seg[i];
    const float* xi = x + (size_t)i * 32;
    const float* hg = h + g * 32;
    float e = 0.f;
#pragma unroll
    for (int j = 0; j < 32; ++j) e += xi[j] * hg[j];
    e_buf[i] = e;
    atomicMaxF(&smax[g], e);
  }
  __syncthreads();
  for (int i = t; i < NG; i += blockDim.x) {
    float v = __int_as_float(smax[i]);
    if (v > -3.0e38f) atomicMaxF((int*)&emax[i], v);
  }
}

__global__ void k_s2s_pass2(const float* __restrict__ x, const int* __restrict__ seg, int n,
                            const float* __restrict__ e_buf, const float* __restrict__ emax,
                            float* __restrict__ denom, float* __restrict__ rsum) {
  __shared__ float sden[NG];
  __shared__ float srs[NG * 32];
  int t = threadIdx.x;
  for (int i = t; i < NG; i += blockDim.x) sden[i] = 0.f;
  for (int i = t; i < NG * 32; i += blockDim.x) srs[i] = 0.f;
  __syncthreads();
  int stride = gridDim.x * blockDim.x;
  for (int i = blockIdx.x * blockDim.x + t; i < n; i += stride) {
    int g = seg[i];
    float em = emax[g];
    if (!(em > -3.0e38f)) em = 0.f;  // where(isfinite(emax), emax, 0)
    float a = __expf(e_buf[i] - em);
    atomicAdd(&sden[g], a);
    const float* xi = x + (size_t)i * 32;
#pragma unroll
    for (int j = 0; j < 32; ++j) atomicAdd(&srs[g * 32 + j], a * xi[j]);
  }
  __syncthreads();
  for (int i = t; i < NG; i += blockDim.x) atomicAdd(&denom[i], sden[i]);
  for (int i = t; i < NG * 32; i += blockDim.x) atomicAdd(&rsum[i], srs[i]);
}

__global__ void k_s2s_final(const float* __restrict__ h, const float* __restrict__ rsum,
                            const float* __restrict__ denom, float* __restrict__ qout) {
  int t = threadIdx.x;
  for (int i = t; i < NG * 64; i += 256) {
    int row = i >> 6, col = i & 63;
    float v = (col < 32) ? h[row * 32 + col]
                         : rsum[row * 32 + (col - 32)] / fmaxf(denom[row], 1e-16f);
    qout[i] = v;
  }
}

// ---------------- head ---------------------------------------------------------

__global__ void k_head(const float* __restrict__ qn, const float* __restrict__ qe,
                       const float* __restrict__ u_resid,
                       const float* W0, const float* B0, const float* W1, const float* B1,
                       const float* Wo, const float* Bo, float* __restrict__ out) {
  int g = threadIdx.x;
  if (g >= NG) return;
  float y[160];
  for (int j = 0; j < 64; ++j) y[j] = qn[g * 64 + j];
  for (int j = 0; j < 64; ++j) y[64 + j] = qe[g * 64 + j];
  for (int j = 0; j < 32; ++j) y[128 + j] = u_resid[g * 32 + j];
  float t1[32], t2[16];
  for (int o = 0; o < 32; ++o) {          // dense has relu_hidden=False
    float a = B0[o];
    for (int j = 0; j < 160; ++j) a += W0[o * 160 + j] * y[j];
    t1[o] = a;
  }
  for (int o = 0; o < 16; ++o) {
    float a = B1[o];
    for (int j = 0; j < 32; ++j) a += W1[o * 32 + j] * t1[j];
    t2[o] = a;
  }
  float a = Bo[0];
  for (int j = 0; j < 16; ++j) a += Wo[j] * t2[j];
  out[g] = a;
}

// ---------------- launch --------------------------------------------------------

extern "C" void kernel_launch(void* const* d_in, const int* in_sizes, int n_in,
                              void* d_out, int out_size, void* d_ws, size_t ws_size,
                              hipStream_t stream) {
  const float* pos   = (const float*)d_in[0];
  const int*   z     = (const int*)d_in[1];
  const int*   ei    = (const int*)d_in[2];
  const int*   batch = (const int*)d_in[3];
  const float* gf    = (const float*)d_in[4];
  const float* emb   = (const float*)d_in[5];
  const float* geW0 = (const float*)d_in[6],  *geB0 = (const float*)d_in[7];
  const float* geW1 = (const float*)d_in[8],  *geB1 = (const float*)d_in[9];
  const float* geW2 = (const float*)d_in[10], *geB2 = (const float*)d_in[11];
  const float* gnW0 = (const float*)d_in[12], *gnB0 = (const float*)d_in[13];
  const float* gnW1 = (const float*)d_in[14], *gnB1 = (const float*)d_in[15];
  const float* gnW2 = (const float*)d_in[16], *gnB2 = (const float*)d_in[17];
  const float* ggW0 = (const float*)d_in[18], *ggB0 = (const float*)d_in[19];
  const float* ggW1 = (const float*)d_in[20], *ggB1 = (const float*)d_in[21];
  const float* ggW2 = (const float*)d_in[22], *ggB2 = (const float*)d_in[23];
  const float* nWih = (const float*)d_in[24], *nWhh = (const float*)d_in[25];
  const float* nbih = (const float*)d_in[26], *nbhh = (const float*)d_in[27];
  const float* eWih = (const float*)d_in[28], *eWhh = (const float*)d_in[29];
  const float* ebih = (const float*)d_in[30], *ebhh = (const float*)d_in[31];
  const float* dW0  = (const float*)d_in[32], *dB0 = (const float*)d_in[33];
  const float* dW1  = (const float*)d_in[34], *dB1 = (const float*)d_in[35];
  const float* oW   = (const float*)d_in[36], *oB  = (const float*)d_in[37];

  char* ws = (char*)d_ws;
  size_t off = 0;
  auto take = [&](size_t bytes) -> char* {
    char* p = ws + off;
    off = (off + bytes + 255) & ~(size_t)255;
    return p;
  };

  _Float16* hgeW0 = (_Float16*)take(4096 * 2);
  _Float16* hgeW1 = (_Float16*)take(1024 * 2);
  _Float16* hgeW2 = (_Float16*)take(1024 * 2);
  _Float16* hgnW0 = (_Float16*)take(3072 * 2);
  _Float16* hgnW1 = (_Float16*)take(1024 * 2);
  _Float16* hgnW2 = (_Float16*)take(1024 * 2);
  _Float16* uh    = (_Float16*)take((size_t)NG * 32 * 2);
  _Float16* xh    = (_Float16*)take((size_t)N_NODES_C * 32 * 2);

  char* zbeg = ws + off;  // contiguous zero-init region
  float* aggsum = (float*)take((size_t)N_NODES_C * 32 * 4);
  float* deg    = (float*)take((size_t)N_NODES_C * 4);
  float* esum   = (float*)take((size_t)NG * 32 * 4);
  float* ecnt   = (float*)take((size_t)NG * 4);
  float* xsum   = (float*)take((size_t)NG * 32 * 4);
  float* ncnt   = (float*)take((size_t)NG * 4);
  size_t zlen = (size_t)((ws + off) - zbeg);

  float* x_resid = (float*)take((size_t)N_NODES_C * 32 * 4);
  float* e_resid = (float*)take((size_t)N_EDGES_C * 32 * 4);
  int*   eb_buf  = (int*)take((size_t)N_EDGES_C * 4);
  float* e_buf   = (float*)take((size_t)N_EDGES_C * 4);
  float* u_resid = (float*)take((size_t)NG * 32 * 4);

  float* hN = (float*)take(NG * 32 * 4); float* cN = (float*)take(NG * 32 * 4);
  float* rN = (float*)take(NG * 32 * 4); float* dN = (float*)take(NG * 4);
  float* mN = (float*)take(NG * 4);      float* qN = (float*)take(NG * 64 * 4);
  float* hE = (float*)take(NG * 32 * 4); float* cE = (float*)take(NG * 32 * 4);
  float* rE = (float*)take(NG * 32 * 4); float* dE = (float*)take(NG * 4);
  float* mE = (float*)take(NG * 4);      float* qE = (float*)take(NG * 64 * 4);

  hipMemsetAsync(zbeg, 0, zlen, stream);

  k_prep<<<1, 256, 0, stream>>>(geW0, geW1, geW2, gnW0, gnW1, gnW2, gf,
                                hgeW0, hgeW1, hgeW2, hgnW0, hgnW1, hgnW2, uh);
  k_embed<<<2048, 256, 0, stream>>>(emb, z, batch, xh, ncnt);

  k_edge_mlp<<<1280, 256, 0, stream>>>(pos, ei, batch, xh, uh,
                                       hgeW0, hgeW1, hgeW2, geB0, geB1, geB2,
                                       e_resid, aggsum, deg, esum, ecnt, eb_buf);
  k_div<<<2048, 256, 0, stream>>>(aggsum, deg);
  k_node_mlp<<<640, 256, 0, stream>>>(emb, z, batch, aggsum, uh,
                                      hgnW0, hgnW1, hgnW2, gnB0, gnB1, gnB2,
                                      x_resid, xsum);
  k_global_mlp<<<1, 64, 0, stream>>>(gf, xsum, ncnt, esum, ecnt,
                                     ggW0, ggB0, ggW1, ggB1, ggW2, ggB2, u_resid);

  for (int step = 0; step < 3; ++step) {
    k_s2s_lstm<<<1, 256, 0, stream>>>(nWih, nWhh, nbih, nbhh, hN, cN, rN, dN, mN, step);
    k_s2s_pass1<<<1024, 256, 0, stream>>>(x_resid, batch, N_NODES_C, hN, e_buf, mN);
    k_s2s_pass2<<<1024, 256, 0, stream>>>(x_resid, batch, N_NODES_C, e_buf, mN, dN, rN);
  }
  k_s2s_final<<<1, 256, 0, stream>>>(hN, rN, dN, qN);

  for (int step = 0; step < 3; ++step) {
    k_s2s_lstm<<<1, 256, 0, stream>>>(eWih, eWhh, ebih, ebhh, hE, cE, rE, dE, mE, step);
    k_s2s_pass1<<<1024, 256, 0, stream>>>(e_resid, eb_buf, N_EDGES_C, hE, e_buf, mE);
    k_s2s_pass2<<<1024, 256, 0, stream>>>(e_resid, eb_buf, N_EDGES_C, e_buf, mE, dE, rE);
  }
  k_s2s_final<<<1, 256, 0, stream>>>(hE, rE, dE, qE);

  k_head<<<1, 64, 0, stream>>>(qN, qE, u_resid, dW0, dB0, dW1, dB1, oW, oB, (float*)d_out);
}
